// SelfAttention_76278619177344
// MI455X (gfx1250) — matrix operand order repaired and checked by reading, not compiled, not run on hardware
//
#include <hip/hip_runtime.h>
#include <cstdint>
#include <cstddef>

// ---------------------------------------------------------------------------
// Self-attention (B=1, S=4096, H=1024, 16 heads, hd=64) for gfx1250 (MI455X).
// Mixed precision: bf16 WMMA (v_wmma_f32_16x16x32_bf16) with fp32 accumulate;
// RoPE + softmax in fp32. GEMM A-tiles staged via the Tensor Data Mover
// (tensor_load_to_lds + s_wait_tensorcnt).
// ---------------------------------------------------------------------------

#define S_LEN  4096
#define HDIM   1024
#define NHEADS 16
#define HD     64

typedef __attribute__((ext_vector_type(16))) __bf16 v16bf;
typedef __attribute__((ext_vector_type(8)))  float  v8f;

union FragU {
    unsigned u[8];
    v16bf    v;
};

__device__ __forceinline__ unsigned short f2bf(float f) {
    unsigned u = __float_as_uint(f);
    unsigned r = u + 0x7FFFu + ((u >> 16) & 1u);   // round-to-nearest-even
    return (unsigned short)(r >> 16);
}

// ---------------------------------------------------------------------------
// Tensor Data Mover: 2D bf16 tile load (global -> LDS).
// D# group0/group1 packed per CDNA5 ISA ch.8 (CDNA5 ISA).
// ---------------------------------------------------------------------------
#ifndef __has_builtin
#define __has_builtin(x) 0
#endif

#if __has_builtin(__builtin_amdgcn_tensor_load_to_lds)
#define USE_TDM 1
#else
#define USE_TDM 0
#endif

#if USE_TDM
typedef __attribute__((ext_vector_type(4))) unsigned tdm_v4u;
typedef __attribute__((ext_vector_type(8))) int      tdm_v8i;
typedef __attribute__((ext_vector_type(4))) int      tdm_v4i;

__device__ __forceinline__ void tdm_load_tile_2d_bf16(
    unsigned lds_off, const void* gptr, unsigned tensor_d0, unsigned tensor_d1,
    unsigned tile_d0, unsigned tile_d1, unsigned stride0) {
    unsigned long long ga = (unsigned long long)gptr;
    tdm_v4u g0;
    g0[0] = 1u;                                        // count=1, user mode
    g0[1] = lds_off;                                   // lds_addr (bytes)
    g0[2] = (unsigned)(ga & 0xFFFFFFFFu);              // global_addr[31:0]
    g0[3] = (unsigned)((ga >> 32) & 0x01FFFFFFu)       // global_addr[56:32]
            | (2u << 30);                              // type=2 ("image")
    tdm_v8i g1;
    g1[0] = (int)(1u << 16);                           // wg_mask=0, data_size=1 (2B)
    g1[1] = (int)((tensor_d0 & 0xFFFFu) << 16);        // tensor_dim0[15:0]
    g1[2] = (int)((tensor_d0 >> 16) |                  // tensor_dim0[31:16]
                  ((tensor_d1 & 0xFFFFu) << 16));      // tensor_dim1[15:0]
    g1[3] = (int)((tensor_d1 >> 16) |                  // tensor_dim1[31:16]
                  ((tile_d0 & 0xFFFFu) << 16));        // tile_dim0
    g1[4] = (int)(tile_d1 & 0xFFFFu);                  // tile_dim1; tile_dim2=0
    g1[5] = (int)stride0;                              // tensor_dim0_stride[31:0]
    g1[6] = 0;                                         // stride0[47:32], dim1_stride lo
    g1[7] = 0;
    tdm_v4i gz = {0, 0, 0, 0};                         // groups 2/3 unused (2D)
#if __clang_major__ >= 23
    tdm_v8i gz8 = {0, 0, 0, 0, 0, 0, 0, 0};
    __builtin_amdgcn_tensor_load_to_lds(g0, g1, gz, gz, gz8, 0);
#else
    __builtin_amdgcn_tensor_load_to_lds(g0, g1, gz, gz, 0);
#endif
}
#endif  // USE_TDM

// ---------------------------------------------------------------------------
// fp32 -> bf16 elementwise conversion
// ---------------------------------------------------------------------------
__global__ void cvt_bf16_kernel(const float* __restrict__ src,
                                unsigned short* __restrict__ dst, int n) {
    int i = blockIdx.x * blockDim.x + threadIdx.x;
    if (i < n) dst[i] = f2bf(src[i]);
}

// ---------------------------------------------------------------------------
// GEMM: C[M,N] (fp32) = A[M,K] (bf16, row-major) x B[K,N] (bf16, row-major)
// Workgroup: 256 threads (8 waves). Tile: 128(M) x 64(N). K-step 32.
// A-tile staged by TDM (wave 0 issues, s_wait_tensorcnt, then barrier);
// B-tile staged transposed by vector path so K-pairs are uint-contiguous.
// ---------------------------------------------------------------------------
__global__ __launch_bounds__(256) void gemm_bf16_wmma(
    const unsigned short* __restrict__ A,
    const unsigned short* __restrict__ B,
    float* __restrict__ C, int M, int N, int K) {
    __shared__ unsigned AsU[128 * 16];   // 128 rows x 32 bf16 (16 uints/row)
    __shared__ unsigned BtU[64 * 16];    // transposed: 64(n) x 32(k) bf16

    const int tid  = threadIdx.x;
    const int wv   = tid >> 5;
    const int lane = tid & 31;
    const int half = lane >> 4;
    const int l16  = lane & 15;
    const int m0   = blockIdx.y * 128;
    const int n0   = blockIdx.x * 64;

    const unsigned* Ag = (const unsigned*)A;
    const unsigned* Bg = (const unsigned*)B;
    const int Ku = K >> 1;
    const int Nu = N >> 1;

    v8f zero = {};
    v8f acc[4] = {zero, zero, zero, zero};

    for (int k0 = 0; k0 < K; k0 += 32) {
        __syncthreads();
#if USE_TDM
        // Stage A tile via Tensor Data Mover: 128x32 bf16 sub-tile of A.
        if (wv == 0) {
            tdm_load_tile_2d_bf16(
                (unsigned)(uintptr_t)&AsU[0],             // flat->LDS: addr[31:0]
                (const void*)(A + (size_t)m0 * K + k0),
                (unsigned)K, (unsigned)M, 32u, 128u, (unsigned)K);
            __builtin_amdgcn_s_wait_tensorcnt(0);
        }
#else
        // Fallback: cooperative vector load of A tile.
        {
            int r  = tid >> 1;
            int c0 = (tid & 1) * 8;
            const unsigned* src = Ag + (size_t)(m0 + r) * Ku + (k0 >> 1);
#pragma unroll
            for (int i = 0; i < 8; ++i) AsU[r * 16 + c0 + i] = src[c0 + i];
        }
#endif
        // Stage B tile transposed: Bt[n][k] so K-pairs are uint-contiguous.
        {
            int r  = tid >> 3;            // k row 0..31
            int c0 = (tid & 7) * 4;       // uint col base
            const unsigned* src = Bg + (size_t)(k0 + r) * Nu + (n0 >> 1);
            if (k0 + 32 < K)
                __builtin_prefetch(src + (size_t)16 * Nu, 0, 0);  // 16 k-rows ahead
            unsigned short* Bt = (unsigned short*)BtU;
#pragma unroll
            for (int i = 0; i < 4; ++i) {
                unsigned v = src[c0 + i];
                int nl = 2 * (c0 + i);
                Bt[(nl + 0) * 32 + r] = (unsigned short)(v & 0xFFFFu);
                Bt[(nl + 1) * 32 + r] = (unsigned short)(v >> 16);
            }
        }
        __syncthreads();

        // A fragment: 16x32 bf16. VGPR j: K=2j+16*half (contiguous uints).
        FragU af;
#pragma unroll
        for (int j = 0; j < 8; ++j)
            af.u[j] = AsU[(wv * 16 + l16) * 16 + j + half * 8];

#pragma unroll
        for (int nt = 0; nt < 4; ++nt) {
            // B fragment: 32x16 bf16. VGPR j: K=2j+16*half (contiguous uints).
            FragU bfr;
#pragma unroll
            for (int j = 0; j < 8; ++j)
                bfr.u[j] = BtU[(nt * 16 + l16) * 16 + j + half * 8];
            acc[nt] = __builtin_amdgcn_wmma_f32_16x16x32_bf16(
                false, af.v, false, bfr.v, (short)0, acc[nt], false, false);
        }
    }

    // C/D layout: VGPR r -> M = r + 8*half, lane&15 -> N.
#pragma unroll
    for (int nt = 0; nt < 4; ++nt)
#pragma unroll
        for (int r = 0; r < 8; ++r)
            C[(size_t)(m0 + wv * 16 + half * 8 + r) * N + n0 + nt * 16 + l16] =
                acc[nt][r];
}

// ---------------------------------------------------------------------------
// RoPE on Q,K (fp32 in -> bf16 out) + plain convert of V.
// Interleaved pair rotation, inv_freq[j] = 10000^(-j/32), angle index d%32.
// One thread per (s, head, pair).
// ---------------------------------------------------------------------------
__global__ void rope_cvt_kernel(const float* __restrict__ Qf,
                                const float* __restrict__ Kf,
                                const float* __restrict__ Vf,
                                unsigned short* __restrict__ Qo,
                                unsigned short* __restrict__ Ko,
                                unsigned short* __restrict__ Vo) {
    int idx = blockIdx.x * blockDim.x + threadIdx.x;
    if (idx >= S_LEN * HDIM / 2) return;
    int s    = idx >> 9;       // / 512 pairs per row
    int p    = idx & 511;
    int head = p >> 5;
    int i    = p & 31;
    int d0   = 2 * i;
    int col0 = head * 64 + d0;
    int col1 = col0 + 1;
    int j0   = d0 & 31;
    int j1   = (d0 + 1) & 31;
    // ln(10000)/32 = 0.28782313
    float a0 = (float)s * expf(-0.28782313f * (float)j0);
    float a1 = (float)s * expf(-0.28782313f * (float)j1);
    float s0, c0, s1, c1;
    sincosf(a0, &s0, &c0);
    sincosf(a1, &s1, &c1);

    size_t base = (size_t)s * HDIM;
    float q0v = Qf[base + col0], q1v = Qf[base + col1];
    Qo[base + col0] = f2bf(q0v * c0 - q1v * s0);
    Qo[base + col1] = f2bf(q1v * c1 + q0v * s1);
    float k0v = Kf[base + col0], k1v = Kf[base + col1];
    Ko[base + col0] = f2bf(k0v * c0 - k1v * s0);
    Ko[base + col1] = f2bf(k1v * c1 + k0v * s1);
    Vo[base + col0] = f2bf(Vf[base + col0]);
    Vo[base + col1] = f2bf(Vf[base + col1]);
}

// ---------------------------------------------------------------------------
// Flash-attention: one WG = (head, 64-query tile), 128 threads = 4 waves.
// Wave w handles queries [q0, q0+16). Online softmax over 64-key blocks.
// S = (Q Kt)*0.125 via WMMA (d-contraction, 2 steps of 32);
// O += P V via WMMA (key-contraction, 2 steps of 32), V staged transposed
// in LDS, P converted C-layout -> A-layout through per-wave LDS.
// ---------------------------------------------------------------------------
__global__ __launch_bounds__(128) void attn_wmma(
    const unsigned short* __restrict__ Qb,
    const unsigned short* __restrict__ Kb,
    const unsigned short* __restrict__ Vb,
    const int* __restrict__ mask,
    unsigned short* __restrict__ ctx) {
    __shared__ unsigned VtU[64 * 32];       // Vt[d][k]: 64x64 bf16 transposed
    __shared__ unsigned PU[4 * 16 * 32];    // per-wave P: 16 rows x 64 keys bf16

    const int tid  = threadIdx.x;
    const int wv   = tid >> 5;
    const int lane = tid & 31;
    const int half = lane >> 4;
    const int l16  = lane & 15;
    const int h    = blockIdx.y;
    const int q0   = blockIdx.x * 64 + wv * 16;

    const unsigned* Qg = (const unsigned*)Qb;
    const unsigned* Kg = (const unsigned*)Kb;
    const unsigned* Vg = (const unsigned*)Vb;
    const int RU = HDIM >> 1;   // 512 uints per row

    // Q A-fragments for both 32-wide d-steps (resident for whole key loop).
    FragU qf[2];
#pragma unroll
    for (int ks = 0; ks < 2; ++ks)
#pragma unroll
        for (int j = 0; j < 8; ++j)
            qf[ks].u[j] = Qg[(size_t)(q0 + l16) * RU + h * 32 + ks * 16 +
                             j + half * 8];

    v8f zero = {};
    v8f oacc[4] = {zero, zero, zero, zero};
    float mrow[8], lrow[8];
#pragma unroll
    for (int r = 0; r < 8; ++r) { mrow[r] = -3.0e38f; lrow[r] = 0.0f; }

    for (int kb = 0; kb < S_LEN / 64; ++kb) {
        __syncthreads();   // prior iteration done consuming Vt
        // Stage V block transposed: Vt[d][k].
        {
            int r  = tid >> 1;                 // key row 0..63
            int c0 = (tid & 1) * 16;
            const unsigned* src = Vg + (size_t)(kb * 64 + r) * RU + h * 32;
            unsigned short* Vt = (unsigned short*)VtU;
#pragma unroll
            for (int i = 0; i < 16; ++i) {
                unsigned v = src[c0 + i];
                int d0 = 2 * (c0 + i);
                Vt[(d0 + 0) * 64 + r] = (unsigned short)(v & 0xFFFFu);
                Vt[(d0 + 1) * 64 + r] = (unsigned short)(v >> 16);
            }
        }

        // Scores: S[16q x 64k] = Q Kt (contraction over d=64).
        v8f sacc[4] = {zero, zero, zero, zero};
#pragma unroll
        for (int nt = 0; nt < 4; ++nt) {
#pragma unroll
            for (int ks = 0; ks < 2; ++ks) {
                // B fragment of Kt: B[d][key] = K[key][d]; VGPR j: d=2j+16*half.
                FragU kf;
#pragma unroll
                for (int j = 0; j < 8; ++j)
                    kf.u[j] = Kg[(size_t)(kb * 64 + nt * 16 + l16) * RU +
                                 h * 32 + ks * 16 + j + half * 8];
                sacc[nt] = __builtin_amdgcn_wmma_f32_16x16x32_bf16(
                    false, qf[ks].v, false, kf.v, (short)0, sacc[nt], false, false);
            }
            int key = kb * 64 + nt * 16 + l16;
            float bias = (mask[key] == 0) ? -3.0e38f : 0.0f;
#pragma unroll
            for (int r = 0; r < 8; ++r)
                sacc[nt][r] = sacc[nt][r] * 0.125f + bias;   // 1/sqrt(64)
        }
        __syncthreads();   // Vt fill complete before P@V reads it

        // Online softmax per row (row m = r + 8*half lives in 16 lanes).
        unsigned short* Ps = (unsigned short*)(PU + wv * 16 * 32);
#pragma unroll
        for (int r = 0; r < 8; ++r) {
            float v = sacc[0][r];
            v = fmaxf(v, sacc[1][r]);
            v = fmaxf(v, sacc[2][r]);
            v = fmaxf(v, sacc[3][r]);
            v = fmaxf(v, __shfl_xor(v, 1));
            v = fmaxf(v, __shfl_xor(v, 2));
            v = fmaxf(v, __shfl_xor(v, 4));
            v = fmaxf(v, __shfl_xor(v, 8));
            float mn    = fmaxf(mrow[r], v);
            float alpha = __expf(mrow[r] - mn);
            mrow[r] = mn;
            lrow[r] *= alpha;
#pragma unroll
            for (int dt = 0; dt < 4; ++dt) oacc[dt][r] *= alpha;
            float rs = 0.0f;
#pragma unroll
            for (int nt = 0; nt < 4; ++nt) {
                float p = __expf(sacc[nt][r] - mn);
                rs += p;
                Ps[(half * 8 + r) * 64 + nt * 16 + l16] = f2bf(p);
            }
            rs += __shfl_xor(rs, 1);
            rs += __shfl_xor(rs, 2);
            rs += __shfl_xor(rs, 4);
            rs += __shfl_xor(rs, 8);
            lrow[r] += rs;
        }

        // P A-fragments (same-wave LDS round trip; DS ops in-order per wave).
        const unsigned* Pu = PU + wv * 16 * 32;
        FragU pf[2];
#pragma unroll
        for (int ks = 0; ks < 2; ++ks)
#pragma unroll
            for (int j = 0; j < 8; ++j)
                pf[ks].u[j] = Pu[l16 * 32 + ks * 16 +
                                 j + half * 8];

        // O += P V (contraction over 64 keys).
#pragma unroll
        for (int dt = 0; dt < 4; ++dt) {
#pragma unroll
            for (int ks = 0; ks < 2; ++ks) {
                FragU vf;
#pragma unroll
                for (int j = 0; j < 8; ++j)
                    vf.u[j] = VtU[(dt * 16 + l16) * 32 + ks * 16 + j + half * 8];
                oacc[dt] = __builtin_amdgcn_wmma_f32_16x16x32_bf16(
                    false, pf[ks].v, false, vf.v, (short)0, oacc[dt], false, false);
            }
        }
    }

    // Normalize and store ctx (bf16) back in [S][H] layout.
#pragma unroll
    for (int dt = 0; dt < 4; ++dt)
#pragma unroll
        for (int r = 0; r < 8; ++r) {
            float val = oacc[dt][r] / lrow[r];
            ctx[(size_t)(q0 + half * 8 + r) * HDIM + h * 64 + dt * 16 + l16] =
                f2bf(val);
        }
}

// ---------------------------------------------------------------------------
// Launch
// ---------------------------------------------------------------------------
extern "C" void kernel_launch(void* const* d_in, const int* in_sizes, int n_in,
                              void* d_out, int out_size, void* d_ws,
                              size_t ws_size, hipStream_t stream) {
    (void)in_sizes; (void)n_in; (void)out_size; (void)ws_size;
    const float* x    = (const float*)d_in[0];
    const float* Wq   = (const float*)d_in[1];
    const float* Wk   = (const float*)d_in[2];
    const float* Wv   = (const float*)d_in[3];
    const float* Wo   = (const float*)d_in[4];
    const int*   mask = (const int*)d_in[5];
    float* out = (float*)d_out;

    char* ws = (char*)d_ws;
    const size_t MB = (size_t)1 << 20;
    unsigned short* xb   = (unsigned short*)(ws + 0 * MB);    // 8 MB
    unsigned short* Wqb  = (unsigned short*)(ws + 8 * MB);    // 2 MB
    unsigned short* Wkb  = (unsigned short*)(ws + 10 * MB);   // 2 MB
    unsigned short* Wvb  = (unsigned short*)(ws + 12 * MB);   // 2 MB
    unsigned short* Wob  = (unsigned short*)(ws + 14 * MB);   // 2 MB
    float*          Qf   = (float*)(ws + 16 * MB);            // 16 MB
    float*          Kf   = (float*)(ws + 32 * MB);            // 16 MB
    float*          Vf   = (float*)(ws + 48 * MB);            // 16 MB
    unsigned short* Qbb  = (unsigned short*)(ws + 64 * MB);   // 8 MB
    unsigned short* Kbb  = (unsigned short*)(ws + 72 * MB);   // 8 MB
    unsigned short* Vbb  = (unsigned short*)(ws + 80 * MB);   // 8 MB
    unsigned short* ctxb = (unsigned short*)(ws + 88 * MB);   // 8 MB

    const int nx = S_LEN * HDIM;   // 4,194,304
    const int nw = HDIM * HDIM;    // 1,048,576

    cvt_bf16_kernel<<<(nx + 255) / 256, 256, 0, stream>>>(x, xb, nx);
    cvt_bf16_kernel<<<(nw + 255) / 256, 256, 0, stream>>>(Wq, Wqb, nw);
    cvt_bf16_kernel<<<(nw + 255) / 256, 256, 0, stream>>>(Wk, Wkb, nw);
    cvt_bf16_kernel<<<(nw + 255) / 256, 256, 0, stream>>>(Wv, Wvb, nw);
    cvt_bf16_kernel<<<(nw + 255) / 256, 256, 0, stream>>>(Wo, Wob, nw);

    dim3 gg(HDIM / 64, S_LEN / 128);   // (16, 32)
    gemm_bf16_wmma<<<gg, 256, 0, stream>>>(xb, Wqb, Qf, S_LEN, HDIM, HDIM);
    gemm_bf16_wmma<<<gg, 256, 0, stream>>>(xb, Wkb, Kf, S_LEN, HDIM, HDIM);
    gemm_bf16_wmma<<<gg, 256, 0, stream>>>(xb, Wvb, Vf, S_LEN, HDIM, HDIM);

    rope_cvt_kernel<<<(nx / 2 + 255) / 256, 256, 0, stream>>>(Qf, Kf, Vf, Qbb,
                                                              Kbb, Vbb);

    dim3 ga(S_LEN / 64, NHEADS);       // (64, 16)
    attn_wmma<<<ga, 128, 0, stream>>>(Qbb, Kbb, Vbb, mask, ctxb);

    gemm_bf16_wmma<<<gg, 256, 0, stream>>>(ctxb, Wob, out, S_LEN, HDIM, HDIM);
}
